// PairwiseConstrainedClustering_49821620633822
// MI455X (gfx1250) — matrix operand order repaired and checked by reading, not compiled, not run on hardware
//
#include <hip/hip_runtime.h>
#include <hip/hip_bf16.h>

typedef __attribute__((ext_vector_type(16))) _Float16 v16h;
typedef __attribute__((ext_vector_type(8)))  _Float16 v8h;
typedef __attribute__((ext_vector_type(8)))  float    v8f;

#define KDIM 32
#define DDIM 256
#define NSEL 4096

__device__ __forceinline__ unsigned ordkey(float f) {
    unsigned u = __float_as_uint(f);
    return (u & 0x80000000u) ? ~u : (u | 0x80000000u);
}

__device__ __forceinline__ float gumbel_noise(unsigned e, unsigned seed) {
    unsigned x = e * 747796405u + seed;
    x ^= x >> 17; x *= 0xed5ad4bbu;
    x ^= x >> 11; x *= 0xac4c1b51u;
    x ^= x >> 15; x *= 0x31848babu;
    x ^= x >> 14;
    float u = (float)(x >> 8) * (1.0f / 16777216.0f);
    u = fmaxf(u, 1e-30f);
    return -logf(-logf(u));
}

// ---------------------------------------------------------------------------
// generic zero
__global__ void zero_u32(unsigned* __restrict__ p, int n) {
    int i = blockIdx.x * blockDim.x + threadIdx.x;
    if (i < n) p[i] = 0u;
}

// ---------------------------------------------------------------------------
// Per-node pass: wave per node, lane = cluster k.
// argmax -> assign, LDS histogram -> counts, sqrt(clusters) -> sqrtc,
// KL partial -> atomic into out (scaled by 1/N).
__global__ void node_kernel(const float* __restrict__ clusters,
                            const float* __restrict__ targets,
                            float* __restrict__ sqrtc,
                            int* __restrict__ assign,
                            unsigned* __restrict__ counts,
                            float* __restrict__ out,
                            int N, float invN) {
    __shared__ unsigned lh[KDIM];
    int t = threadIdx.x, wv = t >> 5, lane = t & 31;
    if (t < KDIM) lh[t] = 0u;
    __syncthreads();
    int n = blockIdx.x * 8 + wv;
    if (n < N) {
        long base = (long)n * KDIM + lane;
        float cl = clusters[base];
        float tg = targets[base];
        sqrtc[base] = sqrtf(cl);
        // wave argmax, ties -> lowest index (matches jnp.argmax)
        float m = cl; int mi = lane;
        #pragma unroll
        for (int o = 16; o > 0; o >>= 1) {
            float om = __shfl_xor(m, o, 32);
            int   oi = __shfl_xor(mi, o, 32);
            if (om > m || (om == m && oi < mi)) { m = om; mi = oi; }
        }
        float term = (tg > 0.f)
            ? tg * (logf(fmaxf(tg, 1e-30f)) - logf(cl)) : 0.f;
        #pragma unroll
        for (int o = 16; o > 0; o >>= 1) term += __shfl_xor(term, o, 32);
        if (lane == 0) {
            assign[n] = mi;
            atomicAdd(&lh[mi], 1u);
            atomicAdd(out, term * invN);
        }
    }
    __syncthreads();
    if (t < KDIM) { unsigned v = lh[t]; if (v) atomicAdd(&counts[t], v); }
}

// ---------------------------------------------------------------------------
// Per-edge pass: wave per edge, lane = cluster k.
// bat = sum_k sqrt(c_src)*sqrt(c_dst) via 128B coalesced gathers (L2-resident).
// Produces Gumbel-perturbed log-prob keys for pos and neg distributions.
__global__ void edge_kernel(const int* __restrict__ edges,
                            const float* __restrict__ sqrtc,
                            const int* __restrict__ assign,
                            const unsigned* __restrict__ counts,
                            float* __restrict__ keypos,
                            float* __restrict__ keyneg,
                            int E, float invN) {
    int t = threadIdx.x, wv = t >> 5, lane = t & 31;
    long e = (long)blockIdx.x * 8 + wv;
    if (e >= E) return;
    int src = edges[2 * e], dst = edges[2 * e + 1];
    float p = sqrtc[(long)src * KDIM + lane] * sqrtc[(long)dst * KDIM + lane];
    #pragma unroll
    for (int o = 16; o > 0; o >>= 1) p += __shfl_xor(p, o, 32);
    if (lane == 0) {
        float bat = p;
        int as = assign[src], ad = assign[dst];
        float fs = (float)counts[as] * invN;
        float fd = (float)counts[ad] * invN;
        float kp = -__builtin_inff(), kn = -__builtin_inff();
        if (as == ad) {
            float nf = (fs == 0.f) ? 1.f : fs;
            float pp = bat / nf;
            if (pp > 0.f) kp = logf(fmaxf(pp, 1e-30f)) + gumbel_noise((unsigned)e, 0x85EBCA6Bu);
        } else {
            float nf = 0.5f * (fs + fd); nf = (nf == 0.f) ? 1.f : nf;
            float np = (1.f - bat) / nf;
            if (np > 0.f) kn = logf(fmaxf(np, 1e-30f)) + gumbel_noise((unsigned)e, 0xC2B2AE35u);
        }
        keypos[e] = kp;
        keyneg[e] = kn;
    }
}

// ---------------------------------------------------------------------------
// Exact top-4096 via 2-level radix select on 16-bit digits.
__global__ void hist_hi(const float* __restrict__ keys, unsigned* __restrict__ hist, int E) {
    long e = (long)blockIdx.x * blockDim.x + threadIdx.x;
    if (e < E) atomicAdd(&hist[ordkey(keys[e]) >> 16], 1u);
}

__global__ void hist_lo(const float* __restrict__ keys, unsigned* __restrict__ hist,
                        const unsigned* __restrict__ sel, int E) {
    long e = (long)blockIdx.x * blockDim.x + threadIdx.x;
    if (e < E) {
        unsigned ou = ordkey(keys[e]);
        if ((ou >> 16) == sel[0]) atomicAdd(&hist[ou & 0xFFFFu], 1u);
    }
}

__global__ void find_kernel(const unsigned* __restrict__ hist,
                            unsigned* __restrict__ sel, int phase) {
    __shared__ unsigned part[256];
    int t = threadIdx.x;
    unsigned s = 0;
    for (int j = 0; j < 256; ++j) s += hist[t * 256 + j];
    part[t] = s;
    __syncthreads();
    if (t == 0) {
        unsigned k = (phase == 0) ? (unsigned)NSEL : sel[1];
        if (k == 0) k = 1;
        unsigned cum = 0; int c = 255;
        for (; c > 0; --c) { if (cum + part[c] >= k) break; cum += part[c]; }
        int b = 255;
        for (; b > 0; --b) { unsigned h = hist[c * 256 + b]; if (cum + h >= k) break; cum += h; }
        unsigned bin = (unsigned)(c * 256 + b);
        unsigned rem = k - cum;   // how many ties at `bin` we still need
        if (phase == 0) { sel[0] = bin; sel[1] = rem; }
        else { sel[2] = (sel[0] << 16) | bin; sel[3] = rem; sel[4] = 0u; sel[5] = 0u; }
    }
}

__global__ void compact_kernel(const float* __restrict__ keys,
                               unsigned* __restrict__ sel,
                               int* __restrict__ idxout, int E) {
    long e = (long)blockIdx.x * blockDim.x + threadIdx.x;
    if (e >= E) return;
    unsigned ou = ordkey(keys[e]);
    unsigned T = sel[2], ties = sel[3];
    if (ties > (unsigned)NSEL) ties = (unsigned)NSEL;
    unsigned ngt = (unsigned)NSEL - ties;
    if (ou > T) {
        unsigned s = atomicAdd(&sel[4], 1u);
        if (s < ngt) idxout[s] = (int)e;
    } else if (ou == T) {
        unsigned s = atomicAdd(&sel[5], 1u);
        if (s < ties) idxout[ngt + s] = (int)e;
    }
}

// ---------------------------------------------------------------------------
// Pairwise loss via WMMA: per wave, one 16-edge tile. For each 32-dim chunk,
// stage f16 diff tile row-major in LDS, build A (16x32) and B = A^T (32x16)
// fragments as contiguous ds_load_b128 runs, accumulate Gram matrix with
// v_wmma_f32_16x16x32_f16, take the diagonal = per-edge ||d||^2.
__global__ void loss_pw_kernel(const int* __restrict__ edges,
                               const float* __restrict__ feats,
                               const int* __restrict__ posidx,
                               const int* __restrict__ negidx,
                               float* __restrict__ out, int E) {
    __shared__ __align__(16) _Float16 tile[8][16][40]; // 40-half pitch keeps 16B align
    __shared__ int srcs[8][16], dsts[8][16];
    int t = threadIdx.x, wv = t >> 5, lane = t & 31;
    int w = blockIdx.x * 8 + wv;              // 512 waves total; block is uniformly pos/neg
    bool isPos = (w < NSEL / 16);
    const int* idx = isPos ? posidx : negidx;
    int tb = (isPos ? w : w - NSEL / 16) * 16;

    if (lane < 16) {
        int e = idx[tb + lane];
        if ((unsigned)e >= (unsigned)E) e = 0;  // guard against underfull selection
        srcs[wv][lane] = edges[2 * e];
        dsts[wv][lane] = edges[2 * e + 1];
    }
    __syncthreads();

    v8f c = {};
    int m = lane & 15, h = lane >> 4;
    for (int ch = 0; ch < 8; ++ch) {
        int base = ch * 32;
        __syncthreads();
        #pragma unroll 4
        for (int i = 0; i < 16; ++i) {
            int s = srcs[wv][i], d = dsts[wv][i];
            float f = feats[(long)s * DDIM + base + lane]
                    - feats[(long)d * DDIM + base + lane];
            tile[wv][i][lane] = (_Float16)f;
        }
        __syncthreads();
        // A fragment: row m, halves = K in [h*8, h*8+8) and [16+h*8, 16+h*8+8)
        v8h a0 = *(const v8h*)&tile[wv][m][h * 8];
        v8h a1 = *(const v8h*)&tile[wv][m][16 + h * 8];
        v16h a = __builtin_shufflevector(a0, a1, 0,1,2,3,4,5,6,7,8,9,10,11,12,13,14,15);
        // B = A^T fragment: column n=m, K in [h*8, h*8+8) and [16+h*8, 16+h*8+8)
        v8h b0 = *(const v8h*)&tile[wv][m][h * 8];
        v8h b1 = *(const v8h*)&tile[wv][m][16 + h * 8];
        v16h b = __builtin_shufflevector(b0, b1, 0,1,2,3,4,5,6,7,8,9,10,11,12,13,14,15);
        c = __builtin_amdgcn_wmma_f32_16x16x32_f16(false, a, false, b,
                                                   (short)0, c, false, false);
    }

    // Diagonal of 16x16 f32 C/D: i<8 -> VGPR i @ lane i ; i>=8 -> VGPR i-8 @ lane i+16
    float dsum = 0.f;
    #pragma unroll
    for (int j = 0; j < 8; ++j) {
        dsum = (lane == j)      ? c[j] : dsum;
        dsum = (lane == j + 24) ? c[j] : dsum;
    }
    #pragma unroll
    for (int o = 16; o > 0; o >>= 1) dsum += __shfl_xor(dsum, o, 32);
    if (lane == 0)
        atomicAdd(out, (isPos ? 1.f : -1.f) * dsum * (1.0f / (float)NSEL));
}

// ---------------------------------------------------------------------------
extern "C" void kernel_launch(void* const* d_in, const int* in_sizes, int n_in,
                              void* d_out, int out_size, void* d_ws, size_t ws_size,
                              hipStream_t stream) {
    const int*   edges    = (const int*)d_in[0];
    const float* feats    = (const float*)d_in[1];
    const float* clusters = (const float*)d_in[2];
    const float* targets  = (const float*)d_in[3];
    float* out = (float*)d_out;

    int E = in_sizes[0] / 2;
    int N = in_sizes[2] / KDIM;
    float invN = 1.0f / (float)N;

    char* ws = (char*)d_ws;
    size_t o = 0;
    float*    sqrtc  = (float*)(ws + o);    o += (size_t)N * KDIM * 4;
    int*      assign = (int*)(ws + o);      o += (size_t)N * 4;
    unsigned* counts = (unsigned*)(ws + o); o += KDIM * 4;
    float*    keypos = (float*)(ws + o);    o += (size_t)E * 4;
    float*    keyneg = (float*)(ws + o);    o += (size_t)E * 4;
    unsigned* hist   = (unsigned*)(ws + o); o += 65536u * 4;
    unsigned* selp   = (unsigned*)(ws + o); o += 32;
    unsigned* seln   = (unsigned*)(ws + o); o += 32;
    int*      posidx = (int*)(ws + o);      o += (size_t)NSEL * 4;
    int*      negidx = (int*)(ws + o);      o += (size_t)NSEL * 4;

    zero_u32<<<1, 64, 0, stream>>>((unsigned*)out, 1);
    zero_u32<<<1, 64, 0, stream>>>(counts, KDIM);

    node_kernel<<<(N + 7) / 8, 256, 0, stream>>>(clusters, targets, sqrtc, assign,
                                                 counts, out, N, invN);
    edge_kernel<<<(E + 7) / 8, 256, 0, stream>>>(edges, sqrtc, assign, counts,
                                                 keypos, keyneg, E, invN);

    int hb = (E + 255) / 256;
    // positive selection
    zero_u32<<<256, 256, 0, stream>>>(hist, 65536);
    hist_hi<<<hb, 256, 0, stream>>>(keypos, hist, E);
    find_kernel<<<1, 256, 0, stream>>>(hist, selp, 0);
    zero_u32<<<256, 256, 0, stream>>>(hist, 65536);
    hist_lo<<<hb, 256, 0, stream>>>(keypos, hist, selp, E);
    find_kernel<<<1, 256, 0, stream>>>(hist, selp, 1);
    compact_kernel<<<hb, 256, 0, stream>>>(keypos, selp, posidx, E);
    // negative selection
    zero_u32<<<256, 256, 0, stream>>>(hist, 65536);
    hist_hi<<<hb, 256, 0, stream>>>(keyneg, hist, E);
    find_kernel<<<1, 256, 0, stream>>>(hist, seln, 0);
    zero_u32<<<256, 256, 0, stream>>>(hist, 65536);
    hist_lo<<<hb, 256, 0, stream>>>(keyneg, hist, seln, E);
    find_kernel<<<1, 256, 0, stream>>>(hist, seln, 1);
    compact_kernel<<<hb, 256, 0, stream>>>(keyneg, seln, negidx, E);

    // pairwise WMMA loss: 2 * (4096/16) = 512 tiles, 8 waves/block -> 64 blocks
    loss_pw_kernel<<<64, 256, 0, stream>>>(edges, feats, posidx, negidx, out, E);
}
